// LlamaAttention_27822798143575
// MI455X (gfx1250) — matrix operand order repaired and checked by reading, not compiled, not run on hardware
//
#include <hip/hip_runtime.h>

// ---------------- problem dims (compile-time constants) ----------------
#define HIDDEN 4096
#define NHEADS 32
#define NKV    8
#define DH     128
#define SEQ    2048
#define NBATCH 2
#define ROWS   (NBATCH*SEQ)   // 4096 token rows

typedef __attribute__((ext_vector_type(16))) __bf16 v16bf;
typedef __attribute__((ext_vector_type(8)))  float  v8f;
typedef __attribute__((ext_vector_type(4)))  unsigned int u32x4;
typedef __attribute__((ext_vector_type(4)))  int i32x4;
typedef __attribute__((ext_vector_type(8)))  int i32x8;

__device__ __forceinline__ v8f zero_v8f() {
  v8f z = {0.f,0.f,0.f,0.f,0.f,0.f,0.f,0.f};
  return z;
}

__device__ __forceinline__ unsigned lds_offset(const void* p) {
  // generic pointer to LDS: high 32 bits = shared aperture, low 32 = byte offset
  return (unsigned)(unsigned long long)p;
}

// ---- Tensor Data Mover: 2D tile (bf16) global -> LDS -----------------------
// D# per CDNA5 ISA:
//   tile_w contiguous elements per row, tile_h rows, row stride in elements.
// LDS receive order: X innermost -> row-major [tile_h][tile_w] at lds_off.
__device__ __forceinline__ void tdm_load_2d_bf16(unsigned lds_off, const void* gptr,
                                                 unsigned tile_w, unsigned tile_h,
                                                 unsigned row_stride,
                                                 unsigned tensor_w, unsigned tensor_h) {
  unsigned long long ga = (unsigned long long)gptr;
  u32x4 g0;
  g0[0] = 1u;                                        // count=1 (valid user D#)
  g0[1] = lds_off;                                   // lds_addr (bytes)
  g0[2] = (unsigned)(ga & 0xffffffffull);            // global_addr[31:0]
  g0[3] = (unsigned)((ga >> 32) & 0x01ffffffull)     // global_addr[56:32]
        | (2u << 30);                                // type = 2 ("image")
  i32x8 g1;
  g1[0] = (int)(1u << 16);                           // data_size=1 (2B), mask=0
  g1[1] = (int)((tensor_w & 0xffffu) << 16);         // tensor_dim0[15:0]
  g1[2] = (int)((tensor_w >> 16) | ((tensor_h & 0xffffu) << 16)); // dim0 hi | dim1 lo
  g1[3] = (int)((tensor_h >> 16) | (tile_w << 16));  // dim1 hi | tile_dim0
  g1[4] = (int)(tile_h & 0xffffu);                   // tile_dim1 (tile_dim2=0)
  g1[5] = (int)row_stride;                           // tensor_dim0_stride[31:0]
  g1[6] = 0;                                         // stride hi | dim1_stride lo
  g1[7] = 0;                                         // dim1_stride hi
  i32x4 gz = {0, 0, 0, 0};                           // groups 2/3 unused (2D)
#if defined(__clang_major__) && __clang_major__ >= 23
  i32x8 gz8 = {0, 0, 0, 0, 0, 0, 0, 0};
  __builtin_amdgcn_tensor_load_to_lds(g0, g1, gz, gz, gz8, 0);
#else
  __builtin_amdgcn_tensor_load_to_lds(g0, g1, gz, gz, 0);
#endif
}

// A fragment (16x32 bf16, contiguous k order matching the B loaders): lane = hf*16 + m holds row m;
// element pair j at k = koff + hf*16 + j*2
__device__ __forceinline__ v16bf load_a_frag(const __bf16* rowbase, int koff, int hf) {
  union { v16bf v; __bf16 e[16]; } u;
#pragma unroll
  for (int j = 0; j < 8; ++j) {
    int k = koff + (j << 1) + (hf << 4);
    u.e[2*j]   = rowbase[k];
    u.e[2*j+1] = rowbase[k+1];
  }
  return u.v;
}

// B fragment (32x16 bf16) from an N-major buffer: element(k,n) = base[n*nstride + k].
__device__ __forceinline__ v16bf load_b_frag_nmajor(const __bf16* base, int nstride, int hf, int ln) {
  union { v16bf v; __bf16 e[16]; } u;
  const __bf16* p = base + ln * nstride + (hf << 4);
#pragma unroll
  for (int j = 0; j < 8; ++j) {
    u.e[2*j]   = p[2*j];
    u.e[2*j+1] = p[2*j+1];
  }
  return u.v;
}

// B fragment from a K-major buffer: element(k,n) = base[k*kstride + n]
__device__ __forceinline__ v16bf load_b_frag_kmajor(const __bf16* base, int kstride, int hf, int ln) {
  union { v16bf v; __bf16 e[16]; } u;
#pragma unroll
  for (int j = 0; j < 8; ++j) {
    int k = (hf << 4) + (j << 1);
    u.e[2*j]   = base[k*kstride + ln];
    u.e[2*j+1] = base[(k+1)*kstride + ln];
  }
  return u.v;
}

// ---------------- fp32 -> bf16 convert ----------------
__global__ void cvt_bf16_kernel(const float* __restrict__ in, __bf16* __restrict__ out, long n) {
  long i = (long)blockIdx.x * blockDim.x + threadIdx.x;
  if (i < n) out[i] = (__bf16)in[i];
}

// ---------------- bf16 GEMM:  C[M,N] = A[M,K] @ Bw[N,K]^T  ----------------
// 128x128 block, 8 waves; K panels of 32 staged by TDM, double-buffered.
__global__ __launch_bounds__(256) void gemm_bf16_nt_kernel(
    const __bf16* __restrict__ A, const __bf16* __restrict__ Bw,
    float* __restrict__ C, int M, int N, int K) {
  __shared__ __bf16 sA[2][128*32];
  __shared__ __bf16 sB[2][128*32];
  const int tid = threadIdx.x;
  const int w = tid >> 5, lane = tid & 31;
  const int hf = lane >> 4, ln = lane & 15;
  const int m_base = blockIdx.y * 128;
  const int n_base = blockIdx.x * 128;
  const int wm = w & 3;      // 4 waves along M (32 rows each)
  const int wn = w >> 2;     // 2 waves along N (64 cols each)
  v8f acc[2][4];
#pragma unroll
  for (int mt = 0; mt < 2; ++mt)
#pragma unroll
    for (int nt = 0; nt < 4; ++nt) acc[mt][nt] = zero_v8f();

  const int nkt = K >> 5;     // number of 32-wide K panels

  // prologue: DMA panel 0 into buffer 0 (wave 0 only; TDM ignores EXEC)
  if (w == 0) {
    tdm_load_2d_bf16(lds_offset(&sA[0][0]), A + (size_t)m_base * K,
                     32, 128, (unsigned)K, (unsigned)K, (unsigned)M);
    tdm_load_2d_bf16(lds_offset(&sB[0][0]), Bw + (size_t)n_base * K,
                     32, 128, (unsigned)K, (unsigned)K, (unsigned)N);
    __builtin_amdgcn_s_wait_tensorcnt(0);
  }
  __syncthreads();

  for (int t = 0; t < nkt; ++t) {
    const int cur = t & 1;
    const __bf16* cA = &sA[cur][0];
    const __bf16* cB = &sB[cur][0];
    // overlap: DMA panel t+1 into the other buffer while computing panel t
    if (w == 0 && t + 1 < nkt) {
      const int kb2 = (t + 1) << 5;
      tdm_load_2d_bf16(lds_offset(&sA[cur ^ 1][0]), A + (size_t)m_base * K + kb2,
                       32, 128, (unsigned)K, (unsigned)K, (unsigned)M);
      tdm_load_2d_bf16(lds_offset(&sB[cur ^ 1][0]), Bw + (size_t)n_base * K + kb2,
                       32, 128, (unsigned)K, (unsigned)K, (unsigned)N);
    }
    v16bf afrag[2], bfrag[4];
#pragma unroll
    for (int mt = 0; mt < 2; ++mt)
      afrag[mt] = load_a_frag(cA + (wm*32 + mt*16 + ln)*32, 0, hf);
#pragma unroll
    for (int nt = 0; nt < 4; ++nt)
      bfrag[nt] = load_b_frag_nmajor(cB + (wn*64 + nt*16)*32, 32, hf, ln);
#pragma unroll
    for (int mt = 0; mt < 2; ++mt)
#pragma unroll
      for (int nt = 0; nt < 4; ++nt)
        acc[mt][nt] = __builtin_amdgcn_wmma_f32_16x16x32_bf16(
            false, afrag[mt], false, bfrag[nt], (short)0, acc[mt][nt], false, false);
    __syncthreads();                       // everyone done reading buffer `cur`
    if (w == 0 && t + 1 < nkt)
      __builtin_amdgcn_s_wait_tensorcnt(0);// next panel landed in LDS
    __syncthreads();                       // release all waves onto next panel
  }

#pragma unroll
  for (int mt = 0; mt < 2; ++mt)
#pragma unroll
    for (int nt = 0; nt < 4; ++nt)
#pragma unroll
      for (int r = 0; r < 8; ++r) {
        int m = m_base + wm*32 + mt*16 + r + (hf << 3);
        int n = n_base + wn*64 + nt*16 + ln;
        C[(size_t)m * N + n] = acc[mt][nt][r];
      }
}

// ---------------- RoPE: f32 [B*S, nh*DH] -> bf16 [B, nh, S, DH] ----------------
__global__ void rope_kernel(const float* __restrict__ in, const int* __restrict__ pos_ids,
                            __bf16* __restrict__ outp, int nh, long total) {
  long idx = (long)blockIdx.x * blockDim.x + threadIdx.x;
  if (idx >= total) return;
  int d0 = (int)(idx & 63);
  long rest = idx >> 6;
  int h = (int)(rest % nh);
  long rest2 = rest / nh;
  int s = (int)(rest2 % SEQ);
  int b = (int)(rest2 / SEQ);
  float p = (float)pos_ids[b*SEQ + s];
  float f = powf(10000.0f, -(float)d0 * (1.0f/64.0f));
  float ang = p * f;
  float c = cosf(ang), sn = sinf(ang);
  const float* row = in + (size_t)(b*SEQ + s) * (nh*DH) + h*DH;
  float x1 = row[d0], x2 = row[d0 + 64];
  __bf16* orow = outp + (((size_t)b*nh + h)*SEQ + s) * DH;
  orow[d0]      = (__bf16)(x1*c - x2*sn);
  orow[d0 + 64] = (__bf16)(x2*c + x1*sn);
}

// ---------------- V: f32 [B*S, NKV*DH] -> bf16 [B, NKV, S, DH] ----------------
__global__ void cvt_v_kernel(const float* __restrict__ in, __bf16* __restrict__ outp) {
  int idx = blockIdx.x * blockDim.x + threadIdx.x;   // 0 .. 2^22-1
  int d = idx & 127;
  int h = (idx >> 7) & 7;
  int s = (idx >> 10) & 2047;
  int b = idx >> 21;
  outp[(((size_t)b*NKV + h)*SEQ + s)*DH + d] =
      (__bf16)in[(size_t)(b*SEQ + s)*(NKV*DH) + h*DH + d];
}

// ---------------- flash attention (causal, GQA) ----------------
// grid: (SEQ/128, NHEADS, NBATCH), block 256 (8 waves, each a 16-row Q tile)
__global__ __launch_bounds__(256) void attn_kernel(
    const __bf16* __restrict__ Q,   // [B, NHEADS, S, DH]
    const __bf16* __restrict__ Kc,  // [B, NKV, S, DH]
    const __bf16* __restrict__ Vc,  // [B, NKV, S, DH]
    __bf16* __restrict__ Oc) {      // [B*S, HIDDEN]
  __shared__ __bf16 sK[32*DH];
  __shared__ __bf16 sV[32*DH];
  __shared__ __bf16 sP[8*16*32];
  const int qblk = blockIdx.x;
  const int h = blockIdx.y;
  const int b = blockIdx.z;
  const int kvh = h >> 2;                 // GQA repeat 4
  const int tid = threadIdx.x;
  const int w = tid >> 5, lane = tid & 31;
  const int hf = lane >> 4, ln = lane & 15;
  const __bf16* qbase = Q  + (((size_t)b*NHEADS + h)*SEQ)*DH;
  const __bf16* kbase = Kc + (((size_t)b*NKV + kvh)*SEQ)*DH;
  const __bf16* vbase = Vc + (((size_t)b*NKV + kvh)*SEQ)*DH;
  const int qb = qblk*128 + w*16;         // wave's q rows [qb, qb+15]

  v16bf qf[4];
#pragma unroll
  for (int ks = 0; ks < 4; ++ks)
    qf[ks] = load_a_frag(qbase + (size_t)(qb + ln)*DH, ks*32, hf);

  v8f oacc[8];
#pragma unroll
  for (int t = 0; t < 8; ++t) oacc[t] = zero_v8f();
  float m_i[8], l_i[8];
#pragma unroll
  for (int r = 0; r < 8; ++r) { m_i[r] = -1e30f; l_i[r] = 0.f; }

  const int nkb = qblk*4 + 4;             // key blocks needed by this q block
  const float scale = 0.088388347648318447f;  // 1/sqrt(128)

  for (int kb = 0; kb < nkb; ++kb) {
    const int kk = kb*32;
    // TDM: 32 keys x 128 dims of K and V -> LDS (wave 0 issues; TENSORcnt)
    if (w == 0) {
      tdm_load_2d_bf16(lds_offset(sK), kbase + (size_t)kk*DH, 128, 32, DH, DH, SEQ);
      tdm_load_2d_bf16(lds_offset(sV), vbase + (size_t)kk*DH, 128, 32, DH, DH, SEQ);
      __builtin_amdgcn_s_wait_tensorcnt(0);
    }
    __syncthreads();
    if (kk <= qb + 15) {                  // wave-uniform: EXEC stays all-ones
      // S = Q @ K^T  (16 x 32 scores, two 16x16 WMMA column-tiles)
      v8f s0 = zero_v8f(), s1 = zero_v8f();
#pragma unroll
      for (int ks = 0; ks < 4; ++ks) {
        v16bf bk0 = load_b_frag_nmajor(sK + ks*32,          DH, hf, ln);
        v16bf bk1 = load_b_frag_nmajor(sK + 16*DH + ks*32,  DH, hf, ln);
        s0 = __builtin_amdgcn_wmma_f32_16x16x32_bf16(false, qf[ks], false, bk0, (short)0, s0, false, false);
        s1 = __builtin_amdgcn_wmma_f32_16x16x32_bf16(false, qf[ks], false, bk1, (short)0, s1, false, false);
      }
      // causal mask + online softmax in C-layout (row = r + 8*hf, col = ln / ln+16)
      __bf16* sp = sP + w*(16*32);
#pragma unroll
      for (int r = 0; r < 8; ++r) {
        const int qrow = qb + r + (hf << 3);
        float v0 = s0[r] * scale;
        float v1 = s1[r] * scale;
        if (kk + ln > qrow)      v0 = -1e30f;
        if (kk + 16 + ln > qrow) v1 = -1e30f;
        float mx = fmaxf(v0, v1);
#pragma unroll
        for (int off = 8; off > 0; off >>= 1)
          mx = fmaxf(mx, __shfl_xor(mx, off, 32));
        const float mnew = fmaxf(m_i[r], mx);
        const float corr = __expf(m_i[r] - mnew);
        const float p0 = __expf(v0 - mnew);
        const float p1 = __expf(v1 - mnew);
        float rs = p0 + p1;
#pragma unroll
        for (int off = 8; off > 0; off >>= 1)
          rs += __shfl_xor(rs, off, 32);
        l_i[r] = l_i[r]*corr + rs;
        m_i[r] = mnew;
#pragma unroll
        for (int t = 0; t < 8; ++t) oacc[t][r] *= corr;
        const int mrow = r + (hf << 3);
        sp[mrow*32 + ln]      = (__bf16)p0;
        sp[mrow*32 + 16 + ln] = (__bf16)p1;
      }
      // transpose P through per-wave LDS patch (wave-private; DS ops in-order per wave)
      v16bf ap = load_a_frag(sp + ln*32, 0, hf);
      // O += P @ V  (K = 32 keys, N = 128 dims -> 8 WMMAs)
#pragma unroll
      for (int t = 0; t < 8; ++t) {
        v16bf bv = load_b_frag_kmajor(sV + t*16, DH, hf, ln);
        oacc[t] = __builtin_amdgcn_wmma_f32_16x16x32_bf16(false, ap, false, bv, (short)0, oacc[t], false, false);
      }
    }
    __syncthreads();                      // before TDM overwrites sK/sV
  }

#pragma unroll
  for (int t = 0; t < 8; ++t)
#pragma unroll
    for (int r = 0; r < 8; ++r) {
      const int qrow = qb + r + (hf << 3);
      float o = oacc[t][r] / l_i[r];
      Oc[((size_t)(b*SEQ + qrow))*HIDDEN + h*DH + t*16 + ln] = (__bf16)o;
    }
}

// ---------------- host-side orchestration ----------------
extern "C" void kernel_launch(void* const* d_in, const int* in_sizes, int n_in,
                              void* d_out, int out_size, void* d_ws, size_t ws_size,
                              hipStream_t stream) {
  const float* hidden = (const float*)d_in[0];
  const int*   pos    = (const int*)d_in[1];
  const float* Wq     = (const float*)d_in[2];
  const float* Wk     = (const float*)d_in[3];
  const float* Wv     = (const float*)d_in[4];
  const float* Wo     = (const float*)d_in[5];
  float* out = (float*)d_out;

  char* ws = (char*)d_ws;
  size_t off = 0;
  auto carve = [&](size_t bytes) -> char* {
    char* p = ws + off;
    off += (bytes + 255) & ~(size_t)255;
    return p;
  };
  __bf16* hid_bf = (__bf16*)carve((size_t)ROWS*HIDDEN*2);
  __bf16* wq_bf  = (__bf16*)carve((size_t)HIDDEN*HIDDEN*2);
  __bf16* wk_bf  = (__bf16*)carve((size_t)NKV*DH*HIDDEN*2);
  __bf16* wv_bf  = (__bf16*)carve((size_t)NKV*DH*HIDDEN*2);
  __bf16* wo_bf  = (__bf16*)carve((size_t)HIDDEN*HIDDEN*2);
  float*  qf32   = (float*) carve((size_t)ROWS*HIDDEN*4);
  float*  kf32   = (float*) carve((size_t)ROWS*NKV*DH*4);
  float*  vf32   = (float*) carve((size_t)ROWS*NKV*DH*4);
  __bf16* q_bf   = (__bf16*)carve((size_t)ROWS*HIDDEN*2);
  __bf16* k_bf   = (__bf16*)carve((size_t)ROWS*NKV*DH*2);
  __bf16* v_bf   = (__bf16*)carve((size_t)ROWS*NKV*DH*2);
  __bf16* att_bf = (__bf16*)carve((size_t)ROWS*HIDDEN*2);

  auto cvt = [&](const float* src, __bf16* dst, long n) {
    cvt_bf16_kernel<<<dim3((unsigned)((n + 255) / 256)), dim3(256), 0, stream>>>(src, dst, n);
  };
  cvt(hidden, hid_bf, (long)ROWS*HIDDEN);
  cvt(Wq, wq_bf, (long)HIDDEN*HIDDEN);
  cvt(Wk, wk_bf, (long)NKV*DH*HIDDEN);
  cvt(Wv, wv_bf, (long)NKV*DH*HIDDEN);
  cvt(Wo, wo_bf, (long)HIDDEN*HIDDEN);

  // Q/K/V projections (C = X @ W^T), fp32 results
  gemm_bf16_nt_kernel<<<dim3(HIDDEN/128, ROWS/128), dim3(256), 0, stream>>>(
      hid_bf, wq_bf, qf32, ROWS, HIDDEN, HIDDEN);
  gemm_bf16_nt_kernel<<<dim3((NKV*DH)/128, ROWS/128), dim3(256), 0, stream>>>(
      hid_bf, wk_bf, kf32, ROWS, NKV*DH, HIDDEN);
  gemm_bf16_nt_kernel<<<dim3((NKV*DH)/128, ROWS/128), dim3(256), 0, stream>>>(
      hid_bf, wv_bf, vf32, ROWS, NKV*DH, HIDDEN);

  // RoPE + head-major relayout to bf16
  long nq = (long)NBATCH*SEQ*NHEADS*64;
  rope_kernel<<<dim3((unsigned)((nq + 255) / 256)), dim3(256), 0, stream>>>(qf32, pos, q_bf, NHEADS, nq);
  long nk = (long)NBATCH*SEQ*NKV*64;
  rope_kernel<<<dim3((unsigned)((nk + 255) / 256)), dim3(256), 0, stream>>>(kf32, pos, k_bf, NKV, nk);
  long nv = (long)NBATCH*SEQ*NKV*DH;
  cvt_v_kernel<<<dim3((unsigned)((nv + 255) / 256)), dim3(256), 0, stream>>>(vf32, v_bf);

  // causal flash attention -> [B*S, HIDDEN] bf16
  attn_kernel<<<dim3(SEQ/128, NHEADS, NBATCH), dim3(256), 0, stream>>>(q_bf, k_bf, v_bf, att_bf);

  // output projection -> fp32 d_out
  gemm_bf16_nt_kernel<<<dim3(HIDDEN/128, ROWS/128), dim3(256), 0, stream>>>(
      att_bf, wo_bf, out, ROWS, HIDDEN, HIDDEN);

  (void)in_sizes; (void)n_in; (void)out_size; (void)ws_size;
}
